// GATv2Encoder_24421184045375
// MI455X (gfx1250) — compile-verified, matrix operand order checked by reading, not run on hardware
//
#include <hip/hip_runtime.h>
#include <hip/hip_bf16.h>
#include <math.h>

typedef _Float16 v16h __attribute__((ext_vector_type(16)));
typedef float    v8f  __attribute__((ext_vector_type(8)));

// ---------- monotone float <-> uint encoding for atomic max on floats ----------
__device__ __forceinline__ unsigned ordf(float f) {
    unsigned u = __float_as_uint(f);
    return (u & 0x80000000u) ? ~u : (u | 0x80000000u);
}
__device__ __forceinline__ float deordf(unsigned u) {
    unsigned v = (u & 0x80000000u) ? (u & 0x7FFFFFFFu) : ~u;
    return __uint_as_float(v);
}

// =============================================================================
// Pack W[128,128] fp32 -> f16 in per-lane WMMA B-register layout.
// Record r = ((kb*2 + lh)*128 + col): 16 halves, element i covers
//   K = 32*kb + (i<8 ? 8*lh+i : 8+8*lh+i), column col.  32 B per record ->
// the GEMM reads B as ONE contiguous 32-byte vector load per k-step.
// =============================================================================
__global__ void pack_w_k(const float* __restrict__ W, _Float16* __restrict__ Wp)
{
    int idx = blockIdx.x * blockDim.x + threadIdx.x;   // 0 .. 128*128-1
    if (idx >= 128 * 128) return;
    const int r  = idx >> 4;          // record
    const int i  = idx & 15;          // element within record
    const int col = r & 127;
    const int lh  = (r >> 7) & 1;
    const int kb  = r >> 8;
    const int kk  = (i < 8) ? (8 * lh + i) : (8 + 8 * lh + i);
    const int K   = 32 * kb + kk;
    Wp[(size_t)r * 16 + i] = (_Float16)W[(size_t)K * 128 + col];
}

// =============================================================================
// GEMM: Y[M,128] = X[M,128] @ W[128,128] + bias[128]  (f16 WMMA, fp32 accum)
// Block = 256 threads = 8 waves; wave w owns the 16x16 tile (rows of block,
// cols [16w,16w+16)).  K fully unrolled: 4 x v_wmma_f32_16x16x32_f16.
// A: 4 x float4 global loads + v_cvt_pk_f16_f32.  B: 1 x 32B packed load.
// =============================================================================
__global__ __launch_bounds__(256) void gemm128_wmma(
    const float* __restrict__ X, const _Float16* __restrict__ Wp,
    const float* __restrict__ bias, float* __restrict__ Y, int M)
{
    const int wv    = threadIdx.x >> 5;    // column tile 0..7
    const int lane  = threadIdx.x & 31;
    const int lh    = lane >> 4;           // lane half -> K octet select
    const int lm    = lane & 15;
    const int mbase = blockIdx.x * 16;

    int rowA = mbase + lm; if (rowA >= M) rowA = M - 1;   // clamp (tail-safe)
    const int colB = wv * 16 + lm;

    v8f acc = {};
#pragma unroll
    for (int kt = 0; kt < 4; ++kt) {
        const int kb = kt * 32;
        // ---- A: rows of X, two 8-float runs selected by lane half ----
        const float4* pr = (const float4*)(X + (size_t)rowA * 128 + kb);
        const float4 x0 = pr[2 * lh + 0];      // K = kb + 8lh + 0..3
        const float4 x1 = pr[2 * lh + 1];      // K = kb + 8lh + 4..7
        const float4 x2 = pr[4 + 2 * lh + 0];  // K = kb + 16 + 8lh + 0..3
        const float4 x3 = pr[4 + 2 * lh + 1];  // K = kb + 16 + 8lh + 4..7
        v16h a;
        a[0]  = (_Float16)x0.x; a[1]  = (_Float16)x0.y;
        a[2]  = (_Float16)x0.z; a[3]  = (_Float16)x0.w;
        a[4]  = (_Float16)x1.x; a[5]  = (_Float16)x1.y;
        a[6]  = (_Float16)x1.z; a[7]  = (_Float16)x1.w;
        a[8]  = (_Float16)x2.x; a[9]  = (_Float16)x2.y;
        a[10] = (_Float16)x2.z; a[11] = (_Float16)x2.w;
        a[12] = (_Float16)x3.x; a[13] = (_Float16)x3.y;
        a[14] = (_Float16)x3.z; a[15] = (_Float16)x3.w;
        // ---- B: one contiguous packed record ----
        const v16h b = *(const v16h*)(Wp + ((size_t)(kt * 2 + lh) * 128 + colB) * 16);
        acc = __builtin_amdgcn_wmma_f32_16x16x32_f16(
                  false, a, false, b, (short)0, acc, false, false);
    }
    // C/D layout: VGPR r holds row (r + 8*lh), col lm of the tile
    const float bb = bias[colB];
    if (mbase + 16 <= M) {                 // fast path: no per-row guards
#pragma unroll
        for (int r = 0; r < 8; ++r)
            Y[(size_t)(mbase + r + 8 * lh) * 128 + colB] = acc[r] + bb;
    } else {
#pragma unroll
        for (int r = 0; r < 8; ++r) {
            const int m = mbase + r + 8 * lh;
            if (m < M) Y[(size_t)m * 128 + colB] = acc[r] + bb;
        }
    }
}

// =============================================================================
// Init kernels
// =============================================================================
__global__ void init_attn_k(unsigned* __restrict__ amaxU, float* __restrict__ denom, int n)
{
    int i = blockIdx.x * blockDim.x + threadIdx.x;
    if (i < n) { amaxU[i] = 0u; denom[i] = 0.f; }   // ordf(x) > 0 for any finite x
}

__global__ void init_out_k(float* __restrict__ out, const float* __restrict__ bias, int n)
{
    int i = blockIdx.x * blockDim.x + threadIdx.x;
    if (i < n) out[i] = bias[i & 127];
}

// =============================================================================
// Pass 1: per-edge attention logit + running per-(dst,head) max.
// One wave per edge; lane = channel-in-head; loop over 4 heads.
// =============================================================================
__global__ __launch_bounds__(256) void edge_alpha_k(
    const float* __restrict__ xl, const int* __restrict__ ei,
    const float* __restrict__ att, float* __restrict__ alpha,
    unsigned* __restrict__ amaxU, int E)
{
    const int wid  = (int)((blockIdx.x * blockDim.x + threadIdx.x) >> 5);
    const int lane = threadIdx.x & 31;
    if (wid >= E) return;
    const int s = ei[wid], d = ei[E + wid];
    const float* ps = xl + (size_t)s * 128;
    const float* pd = xl + (size_t)d * 128;
#pragma unroll
    for (int h = 0; h < 4; ++h) {
        const int c = h * 32 + lane;
        float v = ps[c] + pd[c];
        v = v > 0.f ? v : 0.2f * v;                 // leaky_relu(0.2)
        float p = v * att[c];
#pragma unroll
        for (int off = 16; off >= 1; off >>= 1) p += __shfl_xor(p, off, 32);
        if (lane == 0) {
            alpha[(size_t)wid * 4 + h] = p;
            atomicMax(&amaxU[(size_t)d * 4 + h], ordf(p));
        }
    }
}

// =============================================================================
// Pass 2: alpha <- exp(alpha - max); denom[dst,h] += alpha
// =============================================================================
__global__ void edge_expsum_k(
    float* __restrict__ alpha, const int* __restrict__ ei,
    const unsigned* __restrict__ amaxU, float* __restrict__ denom, int E)
{
    int i = blockIdx.x * blockDim.x + threadIdx.x;
    if (i >= E * 4) return;
    const int e = i >> 2, h = i & 3;
    const int d = ei[E + e];
    const float m  = deordf(amaxU[(size_t)d * 4 + h]);
    const float ex = __expf(alpha[i] - m);
    alpha[i] = ex;
    atomicAdd(&denom[(size_t)d * 4 + h], ex);
}

// =============================================================================
// Pass 3: out[dst] += (alpha/denom) * xl[src]   (one wave per edge)
// =============================================================================
__global__ __launch_bounds__(256) void edge_scatter_k(
    const float* __restrict__ xl, const int* __restrict__ ei,
    const float* __restrict__ alpha, const float* __restrict__ denom,
    float* __restrict__ out, int E)
{
    const int wid  = (int)((blockIdx.x * blockDim.x + threadIdx.x) >> 5);
    const int lane = threadIdx.x & 31;
    if (wid >= E) return;
    const int s = ei[wid], d = ei[E + wid];
    const float* ps = xl + (size_t)s * 128;
    float* po = out + (size_t)d * 128;
#pragma unroll
    for (int h = 0; h < 4; ++h) {
        const float coef = alpha[(size_t)wid * 4 + h] / denom[(size_t)d * 4 + h];
        const int c = h * 32 + lane;
        atomicAdd(&po[c], coef * ps[c]);
    }
}

// =============================================================================
// Exact GELU (matches jax.nn.gelu approximate=False)
// =============================================================================
__global__ void gelu_k(float* __restrict__ x, int n)
{
    int i = blockIdx.x * blockDim.x + threadIdx.x;
    if (i < n) {
        float v = x[i];
        x[i] = 0.5f * v * (1.f + erff(v * 0.70710678118654752440f));
    }
}

// =============================================================================
// LayerNorm over D=128: one wave per row, lane covers 4 channels
// =============================================================================
__global__ __launch_bounds__(256) void layernorm_k(
    const float* __restrict__ y, const float* __restrict__ gamma,
    const float* __restrict__ beta, float* __restrict__ out, int rows)
{
    const int wv = threadIdx.x >> 5, lane = threadIdx.x & 31;
    const int row = blockIdx.x * 8 + wv;
    if (row >= rows) return;
    const float* p = y + (size_t)row * 128 + lane * 4;
    float v0 = p[0], v1 = p[1], v2 = p[2], v3 = p[3];
    float s = v0 + v1 + v2 + v3;
#pragma unroll
    for (int off = 16; off >= 1; off >>= 1) s += __shfl_xor(s, off, 32);
    const float mu = s * (1.f / 128.f);
    const float d0 = v0 - mu, d1 = v1 - mu, d2 = v2 - mu, d3 = v3 - mu;
    float q = d0 * d0 + d1 * d1 + d2 * d2 + d3 * d3;
#pragma unroll
    for (int off = 16; off >= 1; off >>= 1) q += __shfl_xor(q, off, 32);
    const float rs = rsqrtf(q * (1.f / 128.f) + 1e-12f);
    const int c = lane * 4;
    float* o = out + (size_t)row * 128 + c;
    o[0] = d0 * rs * gamma[c + 0] + beta[c + 0];
    o[1] = d1 * rs * gamma[c + 1] + beta[c + 1];
    o[2] = d2 * rs * gamma[c + 2] + beta[c + 2];
    o[3] = d3 * rs * gamma[c + 3] + beta[c + 3];
}

// =============================================================================
// Launch: conv0 -> gelu -> conv1 -> head GEMM -> layernorm
// =============================================================================
extern "C" void kernel_launch(void* const* d_in, const int* in_sizes, int n_in,
                              void* d_out, int out_size, void* d_ws, size_t ws_size,
                              hipStream_t stream)
{
    const float* embs  = (const float*)d_in[0];
    const float* w0    = (const float*)d_in[1];
    const float* b0    = (const float*)d_in[2];
    const float* att0  = (const float*)d_in[3];
    const float* bias0 = (const float*)d_in[4];
    const float* w1    = (const float*)d_in[5];
    const float* b1    = (const float*)d_in[6];
    const float* att1  = (const float*)d_in[7];
    const float* bias1 = (const float*)d_in[8];
    const float* wout  = (const float*)d_in[9];
    const float* bout  = (const float*)d_in[10];
    const float* gamma = (const float*)d_in[11];
    const float* beta  = (const float*)d_in[12];
    const int*   ei0   = (const int*)d_in[13];
    const int*   ei1   = (const int*)d_in[14];

    const int N  = in_sizes[0] / 128;
    const int E  = in_sizes[13] / 2;
    const int BS = out_size / 128;

    // workspace carve-out (256B aligned)
    char* wsp = (char*)d_ws;
    size_t off = 0;
    auto carve = [&](size_t bytes) -> void* {
        void* p = wsp + off; off += (bytes + 255) & ~(size_t)255; return p;
    };
    float*     xl    = (float*)    carve((size_t)N * 128 * 4);  // transformed features
    float*     buf   = (float*)    carve((size_t)N * 128 * 4);  // layer in/out (ping-pong)
    float*     alpha = (float*)    carve((size_t)E * 4 * 4);
    unsigned*  amaxU = (unsigned*) carve((size_t)N * 4 * 4);
    float*     denom = (float*)    carve((size_t)N * 4 * 4);
    float*     ybuf  = (float*)    carve((size_t)BS * 128 * 4);
    _Float16*  Wp    = (_Float16*) carve((size_t)128 * 128 * 2); // packed weights

    const int gemmBlocksN = (N + 15) / 16;
    const int edgeWaves   = (E + 7) / 8;           // 8 waves / 256-thread block
    const int ehBlocks    = (E * 4 + 255) / 256;
    const int attnBlocks  = (N * 4 + 255) / 256;
    const int featBlocks  = (N * 128 + 255) / 256;
    const int packBlocks  = (128 * 128 + 255) / 256;

    // ---- layer 0 ----
    pack_w_k    <<<packBlocks, 256, 0, stream>>>(w0, Wp);
    gemm128_wmma<<<gemmBlocksN, 256, 0, stream>>>(embs, Wp, b0, xl, N);
    init_attn_k <<<attnBlocks, 256, 0, stream>>>(amaxU, denom, N * 4);
    init_out_k  <<<featBlocks, 256, 0, stream>>>(buf, bias0, N * 128);
    edge_alpha_k<<<edgeWaves,  256, 0, stream>>>(xl, ei0, att0, alpha, amaxU, E);
    edge_expsum_k<<<ehBlocks,  256, 0, stream>>>(alpha, ei0, amaxU, denom, E);
    edge_scatter_k<<<edgeWaves,256, 0, stream>>>(xl, ei0, alpha, denom, buf, E);
    gelu_k      <<<featBlocks, 256, 0, stream>>>(buf, N * 128);

    // ---- layer 1 (xl depends only on buf; buf is then free to receive output) ----
    pack_w_k    <<<packBlocks, 256, 0, stream>>>(w1, Wp);
    gemm128_wmma<<<gemmBlocksN, 256, 0, stream>>>(buf, Wp, b1, xl, N);
    init_attn_k <<<attnBlocks, 256, 0, stream>>>(amaxU, denom, N * 4);
    init_out_k  <<<featBlocks, 256, 0, stream>>>(buf, bias1, N * 128);
    edge_alpha_k<<<edgeWaves,  256, 0, stream>>>(xl, ei1, att1, alpha, amaxU, E);
    edge_expsum_k<<<ehBlocks,  256, 0, stream>>>(alpha, ei1, amaxU, denom, E);
    edge_scatter_k<<<edgeWaves,256, 0, stream>>>(xl, ei1, alpha, denom, buf, E);

    // ---- output head: y = buf[:BS] @ wout + bout; layernorm -> d_out ----
    pack_w_k    <<<packBlocks, 256, 0, stream>>>(wout, Wp);
    gemm128_wmma<<<(BS + 15) / 16, 256, 0, stream>>>(buf, Wp, bout, ybuf, BS);
    layernorm_k <<<(BS + 7) / 8,   256, 0, stream>>>(ybuf, gamma, beta, (float*)d_out, BS);
}
